// Selective_Recurrent_Attention_31851477467506
// MI455X (gfx1250) — matrix operand order repaired and checked by reading, not compiled, not run on hardware
//
#include <hip/hip_runtime.h>
#include <hip/hip_bf16.h>
#include <cstddef>
#include <cstdint>

// ---------------------------------------------------------------------------
// Problem constants (from reference)
// ---------------------------------------------------------------------------
static constexpr int BB  = 2;
static constexpr int TT  = 2048;
static constexpr int BT  = BB * TT;          // 4096 rows
static constexpr int DM  = 512;              // d_model
static constexpr int QKD = 256;              // qk dim
static constexpr int VD  = 512;              // v dim
static constexpr int HH  = 8;                // heads
static constexpr int DK  = QKD / HH;         // 32 per-head qk
static constexpr int DV  = VD / HH;          // 64 per-head v
static constexpr float TAU = 16.0f;
static constexpr float SCALE_BASE = 512.0f;

typedef __attribute__((ext_vector_type(16))) __bf16 v16bf;
typedef __attribute__((ext_vector_type(8)))  __bf16 v8bf;
typedef __attribute__((ext_vector_type(8)))  float  v8f;

// ---------------------------------------------------------------------------
// bf16 helpers (bit-level)
// ---------------------------------------------------------------------------
__device__ __forceinline__ __bf16 f2bf(float f) {
  unsigned u = __builtin_bit_cast(unsigned, f);
  unsigned r = (u + 0x7FFFu + ((u >> 16) & 1u)) >> 16;
  unsigned short s = (unsigned short)r;
  return __builtin_bit_cast(__bf16, s);
}
__device__ __forceinline__ float bf2f(__bf16 b) {
  unsigned short s = __builtin_bit_cast(unsigned short, b);
  unsigned u = ((unsigned)s) << 16;
  return __builtin_bit_cast(float, u);
}

__device__ __forceinline__ v8f wmma_bf16(v16bf a, v16bf b, v8f c) {
  return __builtin_amdgcn_wmma_f32_16x16x32_bf16(
      false, a, false, b, (short)0, c, false, false);
}

// ---------------------------------------------------------------------------
// Wide fragment loads (two 16-byte loads per lane instead of 16x u16)
// A-frag 16x32 bf16: lane half selects K banks {8h..8h+7} U {16+8h..16+8h+7}
// B-frag 32x16 bf16 from a K-contiguous row: same K banks {8h..8h+7} U {16+8h..16+8h+7}
// ---------------------------------------------------------------------------
__device__ __forceinline__ v16bf pack16(v8bf lo, v8bf hi) {
  return __builtin_shufflevector(lo, hi, 0, 1, 2, 3, 4, 5, 6, 7,
                                         8, 9, 10, 11, 12, 13, 14, 15);
}
__device__ __forceinline__ v16bf load_afrag(const __bf16* row, int half) {
  v8bf lo = *(const v8bf*)(row + 8 * half);
  v8bf hi = *(const v8bf*)(row + 16 + 8 * half);
  return pack16(lo, hi);
}
__device__ __forceinline__ v16bf load_bfrag(const __bf16* row, int half) {
  v8bf lo = *(const v8bf*)(row + 8 * half);
  v8bf hi = *(const v8bf*)(row + 16 + 8 * half);
  return pack16(lo, hi);
}

// ---------------------------------------------------------------------------
// Optional Tensor Data Mover support (guarded: host pass / older toolchains
// simply compile without it). Arity per bridge doc: clang-22 = 5 args,
// clang-23 = 6 args.
// ---------------------------------------------------------------------------
#if defined(__AMDGCN__) && defined(__gfx1250__) && \
    __has_builtin(__builtin_amdgcn_tensor_load_to_lds) && \
    __has_builtin(__builtin_amdgcn_s_wait_tensorcnt)
#define HAVE_TDM 1
typedef unsigned tdm_v4u __attribute__((ext_vector_type(4)));
typedef int      tdm_v4i __attribute__((ext_vector_type(4)));
typedef int      tdm_v8i __attribute__((ext_vector_type(8)));
#else
#define HAVE_TDM 0
#endif

// ---------------------------------------------------------------------------
// K0a: f32 -> bf16 convert (plain)
// ---------------------------------------------------------------------------
__global__ void cvt_f2bf_kernel(const float* __restrict__ src,
                                __bf16* __restrict__ dst, int n) {
  int i = blockIdx.x * blockDim.x + threadIdx.x;
  if (i < n) dst[i] = f2bf(src[i]);
}

// K0b: f32 [R,C] -> bf16 transposed [C,R] (weights become K-contiguous rows)
__global__ void cvt_f2bf_t_kernel(const float* __restrict__ src,
                                  __bf16* __restrict__ dst, int R, int C) {
  int i = blockIdx.x * blockDim.x + threadIdx.x;
  if (i < R * C) {
    int r = i / C, c = i % C;
    dst[(size_t)c * R + r] = f2bf(src[i]);
  }
}

// ---------------------------------------------------------------------------
// K1: bf16 WMMA GEMM, B pre-transposed:  C[M,N] = A[M,K] @ Bt[N,K]^T
// grid = (M/32, N/64), block = 32 (one wave), 8 wmma per k-step
// ---------------------------------------------------------------------------
__global__ void gemm_bf16_tn(const __bf16* __restrict__ A,
                             const __bf16* __restrict__ Bt,
                             float* __restrict__ C,
                             int M, int N, int K) {
  int lane = threadIdx.x;
  int half = lane >> 4;
  int mr   = lane & 15;
  int tm = blockIdx.x, tn = blockIdx.y;

  const __bf16* arow0 = A + (size_t)(tm * 32 + mr) * K;
  const __bf16* arow1 = A + (size_t)(tm * 32 + 16 + mr) * K;

  v8f acc[2][4] = {};
  for (int k0 = 0; k0 < K; k0 += 32) {
    v16bf a0 = load_afrag(arow0 + k0, half);
    v16bf a1 = load_afrag(arow1 + k0, half);
    __builtin_prefetch(arow0 + k0 + 128, 0, 1);   // global_prefetch_b8
#pragma unroll
    for (int nt = 0; nt < 4; ++nt) {
      const __bf16* brow = Bt + (size_t)(tn * 64 + nt * 16 + mr) * K + k0;
      v16bf b = load_bfrag(brow, half);
      acc[0][nt] = wmma_bf16(a0, b, acc[0][nt]);
      acc[1][nt] = wmma_bf16(a1, b, acc[1][nt]);
    }
  }
#pragma unroll
  for (int sub = 0; sub < 2; ++sub)
#pragma unroll
    for (int nt = 0; nt < 4; ++nt)
#pragma unroll
      for (int r = 0; r < 8; ++r)
        C[(size_t)(tm * 32 + sub * 16 + r + 8 * half) * N +
          tn * 64 + nt * 16 + mr] = acc[sub][nt][r];
}

// ---------------------------------------------------------------------------
// K2: decay logits  g[row,h] = hs[row,:] . W_gamma[:,h] + b_gamma[h]
// ---------------------------------------------------------------------------
__global__ void gamma_kernel(const float* __restrict__ hs,
                             const float* __restrict__ Wg,
                             const float* __restrict__ bg,
                             float* __restrict__ glog) {
  int row = blockIdx.x;
  int h   = threadIdx.y;
  int lane = threadIdx.x;
  float s = 0.f;
  for (int d = lane; d < DM; d += 32)
    s += hs[(size_t)row * DM + d] * Wg[(size_t)d * HH + h];
#pragma unroll
  for (int o = 16; o > 0; o >>= 1) s += __shfl_xor(s, o, 32);
  if (lane == 0) glog[(size_t)row * HH + h] = s + bg[h];
}

// ---------------------------------------------------------------------------
// K3: log-space cumprod scan over T per (b,h); wave32 Hillis-Steele
// ---------------------------------------------------------------------------
__global__ void scan_kernel(const float* __restrict__ glog,
                            float* __restrict__ logb) {
  int bh = blockIdx.x;
  int b = bh / HH, h = bh % HH;
  int lane = threadIdx.x;
  float carry = 0.f;
  for (int t0 = 0; t0 < TT; t0 += 32) {
    int t = t0 + lane;
    float g = glog[((size_t)b * TT + t) * HH + h];
    float v = -log1pf(__expf(-g)) * (1.0f / TAU);
#pragma unroll
    for (int o = 1; o < 32; o <<= 1) {
      float n = __shfl_up(v, o, 32);
      if (lane >= o) v += n;
    }
    v += carry;
    logb[((size_t)b * TT + t) * HH + h] = v;
    carry = __shfl(v, 31, 32);
  }
}

// ---------------------------------------------------------------------------
// K4: xPos rotary + head permutes: q,k -> [bh,T,dk]; kT -> [bh,dk,T];
//     v -> [bh,dv,T] (transposed so B-fragments are K-contiguous)
// grid = BT, block = 512
// ---------------------------------------------------------------------------
__global__ void rotate_kernel(const float* __restrict__ qkv,
                              __bf16* __restrict__ qr,
                              __bf16* __restrict__ kr,
                              __bf16* __restrict__ krT,
                              __bf16* __restrict__ vT) {
  int row = blockIdx.x;                // b*T + t
  int b = row / TT, t = row % TT;
  int tid = threadIdx.x;

  // v: element e in [0,512) -> vT[bh][dv][T]
  {
    int e = tid;
    int h = e >> 6, d = e & 63;
    vT[(((size_t)(b * HH + h)) * DV + d) * TT + t] =
        f2bf(qkv[(size_t)row * 1024 + 512 + e]);
  }

  if (tid < 128) {
    int i = tid;                       // rotary pair
    float posf = (float)t;
    float bs    = (2.0f * i + 0.4f * QKD) / (1.4f * QKD);
    float scale = __powf(bs, posf / SCALE_BASE);
    float invf  = __powf(10000.0f, -(float)i / 128.0f);
    float th = posf * invf;
    float sn = __sinf(th), cs = __cosf(th);

    float q0 = qkv[(size_t)row * 1024 + 2 * i];
    float q1 = qkv[(size_t)row * 1024 + 2 * i + 1];
    float k0 = qkv[(size_t)row * 1024 + QKD + 2 * i];
    float k1 = qkv[(size_t)row * 1024 + QKD + 2 * i + 1];

    float cq = cs * scale, sq = sn * scale;
    float ck = cs / scale, sk = sn / scale;

    int h = (2 * i) >> 5;
    int d = (2 * i) & 31;
    int bh = b * HH + h;
    __bf16 kv0 = f2bf(k0 * ck - k1 * sk);
    __bf16 kv1 = f2bf(k1 * ck + k0 * sk);

    size_t base = ((size_t)bh * TT + t) * DK + d;
    qr[base]     = f2bf(q0 * cq - q1 * sq);
    qr[base + 1] = f2bf(q1 * cq + q0 * sq);
    kr[base]     = kv0;
    kr[base + 1] = kv1;
    krT[((size_t)bh * DK + d) * TT + t]     = kv0;
    krT[((size_t)bh * DK + d + 1) * TT + t] = kv1;
  }
}

// ---------------------------------------------------------------------------
// K5: fused decayed retention (flash-style; T x T never materialized)
// grid = (T/16, B*H), block = 32 (one wave)
// ---------------------------------------------------------------------------
__global__ void retention_kernel(const __bf16* __restrict__ qr,
                                 const __bf16* __restrict__ kr,
                                 const __bf16* __restrict__ vT,
                                 const float*  __restrict__ logb,
                                 const float*  __restrict__ mask,
                                 float* __restrict__ ro) {
  __shared__ __bf16 P[16 * 32];
#if HAVE_TDM
  __shared__ __bf16 KSTAGE[32 * DK];   // TDM staging tile (async prefetch)
#endif
  int lane = threadIdx.x;
  int half = lane >> 4;
  int mr   = lane & 15;
  int ti = blockIdx.x;
  int bh = blockIdx.y;
  int b = bh / HH, h = bh % HH;
  int i0 = ti * 16;
  const float inv_sqrt_dk = 0.17677669529663687f;  // 32^-0.5

#if HAVE_TDM
  {
    // Tensor Data Mover: stage this block's first K tile (32 x dk bf16)
    // into LDS. D# per CDNA5 ISA.
    unsigned long long ga =
        (unsigned long long)(uintptr_t)(kr + (size_t)bh * TT * DK);
    unsigned lds_off = (unsigned)(uintptr_t)(&KSTAGE[0]);
    tdm_v4u g0 = {1u,                                   // count=1, user desc
                  lds_off,                              // lds_addr
                  (unsigned)(ga & 0xFFFFFFFFu),         // global_addr[31:0]
                  (unsigned)((ga >> 32) & 0x1FFFFFFu) | (2u << 30)}; // type=2
    tdm_v8i g1 = {(int)(1u << 16),                      // data_size=2B
                  (int)(32u << 16),                     // tensor_dim0 lo16 = 32
                  (int)(2048u << 16),                   // dim0 hi=0, dim1 lo=2048
                  (int)(32u << 16),                     // dim1 hi=0, tile_dim0=32
                  (int)(32u),                           // tile_dim1=32, tile_dim2=0
                  (int)(32u),                           // tensor_dim0_stride=32
                  0, 0};
    tdm_v4i z4 = {0, 0, 0, 0};
#if __clang_major__ >= 23
    tdm_v8i z8 = {0, 0, 0, 0, 0, 0, 0, 0};
    __builtin_amdgcn_tensor_load_to_lds(g0, g1, z4, z4, z8, 0);
#else
    __builtin_amdgcn_tensor_load_to_lds(g0, g1, z4, z4, 0);
#endif
  }
#endif

  float lbi[8];
#pragma unroll
  for (int r = 0; r < 8; ++r)
    lbi[r] = logb[((size_t)b * TT + i0 + r + 8 * half) * HH + h];

  // Q A-fragment: one row per lane, two b128 loads
  v16bf qa = load_afrag(qr + ((size_t)bh * TT + i0 + mr) * DK, half);

  v8f O[4] = {};
  int nchunks = i0 / 32 + 1;
  for (int jc = 0; jc < nchunks; ++jc) {
    int jbase = jc * 32;

    // S = Q K^T: kr rows are dk-contiguous -> wide B-fragment loads
    v8f S[2] = {};
#pragma unroll
    for (int s = 0; s < 2; ++s) {
      int jcol = jbase + s * 16 + mr;
      v16bf kf = load_bfrag(kr + ((size_t)bh * TT + jcol) * DK, half);
      S[s] = wmma_bf16(qa, kf, S[s]);
    }

    // decay + causal mask, stash P (bf16) in LDS
#pragma unroll
    for (int s = 0; s < 2; ++s) {
      int jg = jbase + s * 16 + mr;
      float lbj = logb[((size_t)b * TT + jg) * HH + h];
      float mj  = mask[(size_t)b * TT + jg];
#pragma unroll
      for (int r = 0; r < 8; ++r) {
        int irow = i0 + r + 8 * half;
        float w = (jg <= irow) ? __expf(lbi[r] - lbj) * mj * inv_sqrt_dk : 0.0f;
        P[(r + 8 * half) * 32 + s * 16 + mr] = f2bf(S[s][r] * w);
      }
    }
    __syncthreads();

    // P as A-fragment: two ds_load_b128 per lane
    v16bf pa = load_afrag(&P[mr * 32], half);

    // O += P V  (vT rows are T-contiguous -> wide B-fragment loads)
#pragma unroll
    for (int nt = 0; nt < 4; ++nt) {
      v16bf vf = load_bfrag(
          vT + ((size_t)bh * DV + nt * 16 + mr) * TT + jbase, half);
      O[nt] = wmma_bf16(pa, vf, O[nt]);
    }
    __syncthreads();
  }

#pragma unroll
  for (int nt = 0; nt < 4; ++nt)
#pragma unroll
    for (int r = 0; r < 8; ++r) {
      int irow = i0 + r + 8 * half;
      ro[((size_t)b * TT + irow) * VD + h * DV + nt * 16 + mr] = O[nt][r];
    }

#if HAVE_TDM
  __builtin_amdgcn_s_wait_tensorcnt(0);   // drain TDM before wave exit
#endif
}

// ---------------------------------------------------------------------------
// K6: group-norm (8 groups of 64) * SiLU(gated) -> bf16 X
// ---------------------------------------------------------------------------
__global__ void gn_gate_kernel(const float* __restrict__ ro,
                               const float* __restrict__ gated,
                               __bf16* __restrict__ X) {
  int row = blockIdx.x;
  int lane = threadIdx.x;
#pragma unroll
  for (int g = 0; g < HH; ++g) {
    size_t base = (size_t)row * VD + g * 64;
    float x0 = ro[base + lane];
    float x1 = ro[base + 32 + lane];
    float s = x0 + x1;
#pragma unroll
    for (int o = 16; o > 0; o >>= 1) s += __shfl_xor(s, o, 32);
    float mu = s * (1.0f / 64.0f);
    float d0 = x0 - mu, d1 = x1 - mu;
    float vv = d0 * d0 + d1 * d1;
#pragma unroll
    for (int o = 16; o > 0; o >>= 1) vv += __shfl_xor(vv, o, 32);
    float rstd = rsqrtf(vv * (1.0f / 64.0f) + 1e-5f);
    float g0 = gated[base + lane];
    float g1 = gated[base + 32 + lane];
    float s0 = g0 / (1.0f + __expf(-g0));
    float s1 = g1 / (1.0f + __expf(-g1));
    X[base + lane]      = f2bf(d0 * rstd * s0);
    X[base + 32 + lane] = f2bf(d1 * rstd * s1);
  }
}

// ---------------------------------------------------------------------------
// K7: curr_kv[b,h] = sum_t (k_t * w_t) v_t^T via WMMA over T
// krT gives dk-major rows (A K-contiguous); vT gives dv-major rows (B).
// ---------------------------------------------------------------------------
__global__ void kv_state_kernel(const __bf16* __restrict__ krT,
                                const __bf16* __restrict__ vT,
                                const float*  __restrict__ logb,
                                const float*  __restrict__ mask,
                                float* __restrict__ out_kv) {
  int lane = threadIdx.x;
  int half = lane >> 4;
  int mr   = lane & 15;
  int bh = blockIdx.x;
  int b = bh / HH, h = bh % HH;
  float lbLast = logb[((size_t)b * TT + (TT - 1)) * HH + h];

  v8f acc[2][4] = {};
  for (int t0 = 0; t0 < TT; t0 += 32) {
    v16bf af[2];
#pragma unroll
    for (int mt = 0; mt < 2; ++mt) {
      const __bf16* krow = krT + ((size_t)bh * DK + mt * 16 + mr) * TT + t0;
      v8bf lo = *(const v8bf*)(krow + 8 * half);
      v8bf hi = *(const v8bf*)(krow + 16 + 8 * half);
#pragma unroll
      for (int j = 0; j < 8; ++j) {
        int ta = t0 + 8 * half + j;
        int tb = t0 + 16 + 8 * half + j;
        float wa = __expf(lbLast - logb[((size_t)b * TT + ta) * HH + h]) *
                   mask[(size_t)b * TT + ta];
        float wb = __expf(lbLast - logb[((size_t)b * TT + tb) * HH + h]) *
                   mask[(size_t)b * TT + tb];
        af[mt][j]     = f2bf(bf2f(lo[j]) * wa);
        af[mt][8 + j] = f2bf(bf2f(hi[j]) * wb);
      }
    }
#pragma unroll
    for (int nt = 0; nt < 4; ++nt) {
      v16bf vf = load_bfrag(vT + ((size_t)bh * DV + nt * 16 + mr) * TT + t0, half);
#pragma unroll
      for (int mt = 0; mt < 2; ++mt)
        acc[mt][nt] = wmma_bf16(af[mt], vf, acc[mt][nt]);
    }
  }
#pragma unroll
  for (int mt = 0; mt < 2; ++mt)
#pragma unroll
    for (int nt = 0; nt < 4; ++nt)
#pragma unroll
      for (int r = 0; r < 8; ++r)
        out_kv[((size_t)bh * DK + mt * 16 + r + 8 * half) * DV + nt * 16 + mr] =
            acc[mt][nt][r];
}

// ---------------------------------------------------------------------------
// Host launch
// ---------------------------------------------------------------------------
extern "C" void kernel_launch(void* const* d_in, const int* in_sizes, int n_in,
                              void* d_out, int out_size, void* d_ws, size_t ws_size,
                              hipStream_t stream) {
  (void)in_sizes; (void)n_in; (void)out_size; (void)ws_size;

  const float* hs    = (const float*)d_in[0];   // [B,T,DM]
  const float* mask  = (const float*)d_in[2];   // [B,T]
  const float* Wqkv  = (const float*)d_in[3];   // [DM, 1024]
  const float* Wg    = (const float*)d_in[4];   // [DM, VD]
  const float* Wp    = (const float*)d_in[5];   // [VD, DM]
  const float* Wgam  = (const float*)d_in[6];   // [DM, H]
  const float* bgam  = (const float*)d_in[7];   // [H]
  float* out = (float*)d_out;                   // [B,T,DM] ++ [B,H,DK,DV]

  char* ws = (char*)d_ws;
  const size_t MB = 1u << 20;
  __bf16* hs_bf    = (__bf16*)(ws + 0 * MB);         // 4 MiB  [BT,DM]
  __bf16* wqkv_t   = (__bf16*)(ws + 4 * MB);         // 1 MiB  [1024,DM]
  __bf16* wg_t     = (__bf16*)(ws + 5 * MB);         // 0.5MiB [VD,DM]
  __bf16* wp_t     = (__bf16*)(ws + 5 * MB + 512 * 1024); // [DM,VD]
  float*  qkv_f    = (float*) (ws + 6 * MB);         // 16 MiB [BT,1024]
  float*  glog     = (float*) (ws + 22 * MB);        // 128 KiB
  float*  logb     = (float*) (ws + 22 * MB + 128 * 1024);
  __bf16* qrot     = (__bf16*)(ws + 23 * MB);        // 2 MiB  [bh,T,dk]
  __bf16* krot     = (__bf16*)(ws + 25 * MB);        // 2 MiB  [bh,T,dk]
  __bf16* vtrans   = (__bf16*)(ws + 27 * MB);        // 4 MiB  [bh,dv,T]
  float*  gated_f  = (float*) (ws + 31 * MB);        // 8 MiB
  float*  ro_f     = (float*) (ws + 39 * MB);        // 8 MiB
  __bf16* x_bf     = (__bf16*)(ws + 47 * MB);        // 4 MiB
  __bf16* krot_t   = (__bf16*)(ws + 51 * MB);        // 2 MiB  [bh,dk,T]

  // 1) bf16 casts; weights transposed to [N,K] for K-contiguous B-fragments
  cvt_f2bf_kernel<<<(BT * DM + 255) / 256, 256, 0, stream>>>(hs, hs_bf, BT * DM);
  cvt_f2bf_t_kernel<<<(DM * 1024 + 255) / 256, 256, 0, stream>>>(Wqkv, wqkv_t, DM, 1024);
  cvt_f2bf_t_kernel<<<(DM * VD + 255) / 256, 256, 0, stream>>>(Wg, wg_t, DM, VD);
  cvt_f2bf_t_kernel<<<(VD * DM + 255) / 256, 256, 0, stream>>>(Wp, wp_t, VD, DM);

  // 2) dense GEMMs
  gemm_bf16_tn<<<dim3(BT / 32, 1024 / 64), 32, 0, stream>>>(
      hs_bf, wqkv_t, qkv_f, BT, 1024, DM);
  gemm_bf16_tn<<<dim3(BT / 32, VD / 64), 32, 0, stream>>>(
      hs_bf, wg_t, gated_f, BT, VD, DM);

  // 3) decay logits + log-space scan
  gamma_kernel<<<BT, dim3(32, 8), 0, stream>>>(hs, Wgam, bgam, glog);
  scan_kernel<<<BB * HH, 32, 0, stream>>>(glog, logb);

  // 4) xPos rotary + permutes
  rotate_kernel<<<BT, 512, 0, stream>>>(qkv_f, qrot, krot, krot_t, vtrans);

  // 5) fused decayed retention
  retention_kernel<<<dim3(TT / 16, BB * HH), 32, 0, stream>>>(
      qrot, krot, vtrans, logb, mask, ro_f);

  // 6) group-norm * SiLU gate
  gn_gate_kernel<<<BT, 32, 0, stream>>>(ro_f, gated_f, x_bf);

  // 7) output projection into d_out
  gemm_bf16_tn<<<dim3(BT / 32, DM / 64), 32, 0, stream>>>(
      x_bf, wp_t, out, BT, DM, VD);

  // 8) recurrent state
  kv_state_kernel<<<BB * HH, 32, 0, stream>>>(
      krot_t, vtrans, logb, mask, out + (size_t)BB * TT * DM);
}
